// RGATLayer_26190710571685
// MI455X (gfx1250) — matrix operand order repaired and checked by reading, not compiled, not run on hardware
//
#include <hip/hip_runtime.h>

#define D  128
#define RR 32
#define BB 8

typedef __attribute__((ext_vector_type(16))) __bf16       bf16x16;
typedef __attribute__((ext_vector_type(8)))  float        f32x8;
typedef __attribute__((ext_vector_type(4)))  float        f32x4;
typedef __attribute__((ext_vector_type(4)))  unsigned int u32x4;

union FragAB { bf16x16 v; u32x4 q[2]; };

__device__ __forceinline__ unsigned short f2bf(float f) {
  unsigned int u = __float_as_uint(f);
  u += 0x7fffu + ((u >> 16) & 1u);   // round-to-nearest-even
  return (unsigned short)(u >> 16);
}

// ---------------------------------------------------------------------------
// Prep: Wt[r][o][i] = bf16( sum_b coef[r][b] * bases[b][i][o] )  (B transposed
// so WMMA B-fragments are contiguous in K), plus Lt[o][i] = bf16(w_loop[i][o]).
// ---------------------------------------------------------------------------
__global__ void build_weights_kernel(const float* __restrict__ coef,
                                     const float* __restrict__ bases,
                                     const float* __restrict__ wloop,
                                     unsigned short* __restrict__ Wt,
                                     unsigned short* __restrict__ Lt) {
  int idx = blockIdx.x * 256 + threadIdx.x;
  int r = idx >> 14;           // 14 = log2(128*128)
  int rem = idx & 16383;
  int o = rem >> 7, i = rem & 127;
  if (r < RR) {
    float s = 0.0f;
#pragma unroll
    for (int b = 0; b < BB; ++b)
      s += coef[r * BB + b] * bases[((size_t)b * D + i) * D + o];
    Wt[((size_t)r * D + o) * D + i] = f2bf(s);
  } else if (r == RR) {
    Lt[(size_t)o * D + i] = f2bf(wloop[(size_t)i * D + o]);
  }
}

__global__ void cvt_x_bf16(const float* __restrict__ x,
                           unsigned short* __restrict__ xb, int n4) {
  int i = blockIdx.x * blockDim.x + threadIdx.x;
  if (i >= n4) return;
  f32x4 v = ((const f32x4*)x)[i];
  unsigned long long pk =
      (unsigned long long)f2bf(v.x) |
      ((unsigned long long)f2bf(v.y) << 16) |
      ((unsigned long long)f2bf(v.z) << 32) |
      ((unsigned long long)f2bf(v.w) << 48);
  ((unsigned long long*)xb)[i] = pk;
}

__global__ void init_kernel(float* accum, float* denom, float* mmax,
                            float* colsum, float* colsumsq, int N) {
  long long i = (long long)blockIdx.x * blockDim.x + threadIdx.x;
  long long tot = (long long)N * D;
  if (i < tot) accum[i] = 0.0f;
  if (i < N) { denom[i] = 0.0f; mmax[i] = -__builtin_huge_valf(); }
  if (i < D) { colsum[i] = 0.0f; colsumsq[i] = 0.0f; }
}

// ---------------------------------------------------------------------------
// Tiled bf16 WMMA GEMM: C tile (128x128) = A(128x128) * Bt^T.
// A: [M][128] bf16 row-major.  Bt: [nrel][128(o)][128(i)] bf16 (pre-transposed
// weight, so both fragments load as two contiguous ds_read_b128).
// Tiles staged via one-hop GLOBAL_LOAD_ASYNC_TO_LDS_B128 (ASYNCcnt-tracked).
// grid = (ceil(M/128), nrel), block = 256 (8 wave32).
// ---------------------------------------------------------------------------
__global__ __launch_bounds__(256) void gemm_wmma_bf16(
    const unsigned short* __restrict__ A,
    const unsigned short* __restrict__ Bt,
    float* __restrict__ C, int M, int ldc, int relStrideC) {
  __shared__ unsigned short sA[128][136];   // +8 shorts pad (16B) per row
  __shared__ unsigned short sB[128][136];

  const int tid = threadIdx.x;
  const int row0 = blockIdx.x * 128;
  const unsigned short* Bp = Bt + (size_t)blockIdx.y * (D * D);
  float* Cb = C + (size_t)blockIdx.y * relStrideC;

  // Stage A tile (EXEC-guarded) and B tile: global -> LDS one-hop async copy,
  // 16 bytes per lane per op. Out-of-range A rows stay stale in LDS but only
  // feed accumulator rows whose stores are guarded below.
  for (int i = tid; i < 128 * 16; i += 256) {
    int r = i >> 4;
    int cq = (i & 15) << 3;                  // column (shorts), multiple of 8
    unsigned lB = (unsigned)(unsigned long long)&sB[r][cq];
    unsigned long long gB = (unsigned long long)(Bp + (size_t)r * D + cq);
    asm volatile("global_load_async_to_lds_b128 %0, %1, off"
                 :: "v"(lB), "v"(gB) : "memory");
    if (row0 + r < M) {
      unsigned lA = (unsigned)(unsigned long long)&sA[r][cq];
      unsigned long long gA =
          (unsigned long long)(A + (size_t)(row0 + r) * D + cq);
      asm volatile("global_load_async_to_lds_b128 %0, %1, off"
                   :: "v"(lA), "v"(gA) : "memory");
    }
  }
  asm volatile("s_wait_asynccnt 0" ::: "memory");
  __syncthreads();

  const int wave = tid >> 5, lane = tid & 31;
  const int l16 = lane & 15, hi = lane >> 4;
  const int m0 = wave << 4;                  // this wave's 16 M-rows

  f32x8 acc[8] = {};

#pragma unroll
  for (int kk = 0; kk < 4; ++kk) {           // K = 4 x 32
    const int k0 = kk << 5;
    // A fragment (16x32 bf16): lane<16 -> K {k0..k0+7, k0+16..k0+23},
    //                          lane>=16 -> K {k0+8..k0+15, k0+24..k0+31}
    FragAB a;
    const unsigned short* ap = &sA[m0 + l16][k0 + hi * 8];
    a.q[0] = *(const u32x4*)ap;
    a.q[1] = *(const u32x4*)(ap + 16);
#pragma unroll
    for (int nt = 0; nt < 8; ++nt) {
      // B fragment (32x16): lane<16 -> col n, K {k0..k0+7, k0+16..k0+23}; lane>=16 -> K+8
      FragAB b;
      const unsigned short* bp = &sB[(nt << 4) + l16][k0 + hi * 8];
      b.q[0] = *(const u32x4*)bp;
      b.q[1] = *(const u32x4*)(bp + 16);
      acc[nt] = __builtin_amdgcn_wmma_f32_16x16x32_bf16(
          false, a.v, false, b.v, (short)0, acc[nt], false, false);
    }
  }

  // C/D layout: VGPR v, lanes 0-15 -> (M=v, N=lane); lanes 16-31 -> (M=v+8).
#pragma unroll
  for (int nt = 0; nt < 8; ++nt) {
    const int col = (nt << 4) + l16;
#pragma unroll
    for (int v = 0; v < 8; ++v) {
      int gr = row0 + m0 + (hi ? 8 : 0) + v;
      if (gr < M) Cb[(size_t)gr * ldc + col] = acc[nt][v];
    }
  }
}

// ---------------------------------------------------------------------------
// Edge phase: one wave per edge; 128-dim dot reduced across 32 lanes.
// ---------------------------------------------------------------------------
__global__ void edge_score_kernel(const float* __restrict__ xr,
                                  const int* __restrict__ src,
                                  const int* __restrict__ dst,
                                  const int* __restrict__ rel,
                                  const float* __restrict__ att,
                                  float* __restrict__ score,
                                  float* __restrict__ mmax, int E) {
  int gw = (blockIdx.x * blockDim.x + threadIdx.x) >> 5;
  int lane = threadIdx.x & 31;
  if (gw >= E) return;
  int s = src[gw], d = dst[gw], r = rel[gw];
  f32x4 sv = ((const f32x4*)(xr + ((size_t)s * RR + r) * D))[lane];
  f32x4 dv = ((const f32x4*)(xr + ((size_t)d * RR + r) * D))[lane];
  f32x4 av = ((const f32x4*)att)[lane];
  float part = (sv.x + dv.x) * av.x + (sv.y + dv.y) * av.y +
               (sv.z + dv.z) * av.z + (sv.w + dv.w) * av.w;
#pragma unroll
  for (int off = 16; off > 0; off >>= 1) part += __shfl_xor(part, off, 32);
  if (lane == 0) {
    float sc = part >= 0.0f ? part : 0.01f * part;   // leaky_relu
    score[gw] = sc;
    // monotone-int trick for float atomic max (init is -inf)
    if (sc >= 0.0f) atomicMax((int*)&mmax[d], __float_as_int(sc));
    else            atomicMin((unsigned int*)&mmax[d], __float_as_uint(sc));
  }
}

__global__ void edge_exp_kernel(const float* __restrict__ score,
                                const int* __restrict__ dst,
                                const float* __restrict__ mmax,
                                float* __restrict__ ex,
                                float* __restrict__ denom, int E) {
  int e = blockIdx.x * blockDim.x + threadIdx.x;
  if (e >= E) return;
  int d = dst[e];
  float v = __expf(score[e] - mmax[d]);   // mmax finite: this edge exists
  ex[e] = v;
  atomicAdd(&denom[d], v);
}

__global__ void edge_accum_kernel(const float* __restrict__ xr,
                                  const int* __restrict__ src,
                                  const int* __restrict__ dst,
                                  const int* __restrict__ rel,
                                  const float* __restrict__ ex,
                                  const float* __restrict__ denom,
                                  float* __restrict__ accum, int E) {
  int gw = (blockIdx.x * blockDim.x + threadIdx.x) >> 5;
  int lane = threadIdx.x & 31;
  if (gw >= E) return;
  int s = src[gw], d = dst[gw], r = rel[gw];
  float c = ex[gw] / denom[d];
  f32x4 sv = ((const f32x4*)(xr + ((size_t)s * RR + r) * D))[lane];
  float* ab = accum + (size_t)d * D + lane * 4;
  atomicAdd(ab + 0, c * sv.x);   // non-returning global_atomic_add_f32
  atomicAdd(ab + 1, c * sv.y);
  atomicAdd(ab + 2, c * sv.z);
  atomicAdd(ab + 3, c * sv.w);
}

// ---------------------------------------------------------------------------
// ft = accum*norm + bias + loop; column sums/sumsq for BatchNorm.
// blockDim = 128: thread c owns column c (coalesced row loads).
// ---------------------------------------------------------------------------
__global__ void finalize_ft_kernel(const float* __restrict__ accum,
                                   const float* __restrict__ norm,
                                   const float* __restrict__ bias,
                                   const float* __restrict__ loopout,
                                   float* __restrict__ ft,
                                   float* __restrict__ colsum,
                                   float* __restrict__ colsumsq, int N) {
  int c = threadIdx.x;
  float s = 0.0f, ss = 0.0f;
  float bc = bias[c];
  for (int r = blockIdx.x; r < N; r += gridDim.x) {
    float v = accum[(size_t)r * D + c] * norm[r] + bc + loopout[(size_t)r * D + c];
    ft[(size_t)r * D + c] = v;
    s += v; ss += v * v;
  }
  atomicAdd(&colsum[c], s);
  atomicAdd(&colsumsq[c], ss);
}

__global__ void bn_stats_kernel(const float* colsum, const float* colsumsq,
                                const float* gamma, const float* beta,
                                float* scale, float* shift, int N) {
  int c = threadIdx.x;
  float inv_n = 1.0f / (float)N;
  float mu = colsum[c] * inv_n;
  float var = colsumsq[c] * inv_n - mu * mu;   // biased, matches reference
  float is = rsqrtf(var + 1e-5f);
  float sc = gamma[c] * is;
  scale[c] = sc;
  shift[c] = beta[c] - mu * sc;
}

__global__ void bn_relu_kernel(float* __restrict__ out,
                               const float* __restrict__ scale,
                               const float* __restrict__ shift, int n4) {
  int i = blockIdx.x * blockDim.x + threadIdx.x;
  if (i >= n4) return;
  f32x4 v = ((const f32x4*)out)[i];
  int cb = (i & 31) << 2;
  v.x = fmaxf(0.0f, v.x * scale[cb + 0] + shift[cb + 0]);
  v.y = fmaxf(0.0f, v.y * scale[cb + 1] + shift[cb + 1]);
  v.z = fmaxf(0.0f, v.z * scale[cb + 2] + shift[cb + 2]);
  v.w = fmaxf(0.0f, v.w * scale[cb + 3] + shift[cb + 3]);
  ((f32x4*)out)[i] = v;
}

// ---------------------------------------------------------------------------
extern "C" void kernel_launch(void* const* d_in, const int* in_sizes, int n_in,
                              void* d_out, int out_size, void* d_ws, size_t ws_size,
                              hipStream_t stream) {
  const float* x      = (const float*)d_in[0];
  const float* norm   = (const float*)d_in[1];
  const int*   esrc   = (const int*)d_in[2];
  const int*   edst   = (const int*)d_in[3];
  const int*   erel   = (const int*)d_in[4];
  const float* wloop  = (const float*)d_in[5];
  const float* wbases = (const float*)d_in[6];
  const float* wcoef  = (const float*)d_in[7];
  const float* att    = (const float*)d_in[8];
  const float* wbias  = (const float*)d_in[9];
  const float* gamma  = (const float*)d_in[10];
  const float* beta   = (const float*)d_in[11];
  float* out = (float*)d_out;

  const int N = in_sizes[1];   // norm has N elements
  const int E = in_sizes[2];   // edge_src has E elements

  char* p = (char*)d_ws;
  auto carve = [&](size_t bytes) -> char* {
    char* r = p;
    p += (bytes + 255) & ~(size_t)255;
    return r;
  };
  float*          xr      = (float*)carve((size_t)N * RR * D * 4);  // 655 MB
  unsigned short* xb      = (unsigned short*)carve((size_t)N * D * 2);
  unsigned short* Wt      = (unsigned short*)carve((size_t)RR * D * D * 2);
  unsigned short* Lt      = (unsigned short*)carve((size_t)D * D * 2);
  float*          loopout = (float*)carve((size_t)N * D * 4);
  float*          score   = (float*)carve((size_t)E * 4);
  float*          ex      = (float*)carve((size_t)E * 4);
  float*          mmax    = (float*)carve((size_t)N * 4);
  float*          denom   = (float*)carve((size_t)N * 4);
  float*          accum   = (float*)carve((size_t)N * D * 4);
  float*          colsum  = (float*)carve(D * 4);
  float*          colss   = (float*)carve(D * 4);
  float*          scale   = (float*)carve(D * 4);
  float*          shift   = (float*)carve(D * 4);

  build_weights_kernel<<<((RR + 1) * D * D + 255) / 256, 256, 0, stream>>>(
      wcoef, wbases, wloop, Wt, Lt);
  cvt_x_bf16<<<(N * D / 4 + 255) / 256, 256, 0, stream>>>(x, xb, N * D / 4);
  init_kernel<<<(N * D + 255) / 256, 256, 0, stream>>>(accum, denom, mmax,
                                                       colsum, colss, N);

  dim3 gX((N + 127) / 128, RR);
  gemm_wmma_bf16<<<gX, 256, 0, stream>>>(xb, Wt, xr, N, RR * D, D);
  dim3 gL((N + 127) / 128, 1);
  gemm_wmma_bf16<<<gL, 256, 0, stream>>>(xb, Lt, loopout, N, D, 0);

  edge_score_kernel<<<(E * 32 + 255) / 256, 256, 0, stream>>>(
      xr, esrc, edst, erel, att, score, mmax, E);
  edge_exp_kernel<<<(E + 255) / 256, 256, 0, stream>>>(score, edst, mmax, ex,
                                                       denom, E);
  edge_accum_kernel<<<(E * 32 + 255) / 256, 256, 0, stream>>>(
      xr, esrc, edst, erel, ex, denom, accum, E);

  finalize_ft_kernel<<<512, 128, 0, stream>>>(accum, norm, wbias, loopout, out,
                                              colsum, colss, N);
  bn_stats_kernel<<<1, 128, 0, stream>>>(colsum, colss, gamma, beta, scale,
                                         shift, N);
  bn_relu_kernel<<<(N * 32 + 255) / 256, 256, 0, stream>>>(out, scale, shift,
                                                           N * 32);
}
